// MultiheadAttentionBlock_44968307589139
// MI455X (gfx1250) — matrix operand order repaired and checked by reading, not compiled, not run on hardware
//
#include <hip/hip_runtime.h>
#include <hip/hip_bf16.h>

typedef __attribute__((ext_vector_type(16))) __bf16 v16bf;
typedef __attribute__((ext_vector_type(8)))  __bf16 v8bf;
typedef __attribute__((ext_vector_type(8)))  float  v8f;

union V16 { v16bf v; v8bf h[2]; };

__device__ __forceinline__ __bf16 f2bf(float f) {
    unsigned u = __builtin_bit_cast(unsigned, f);
    unsigned r = (u + 0x7fffu + ((u >> 16) & 1u)) >> 16;
    unsigned short s = (unsigned short)r;
    return __builtin_bit_cast(__bf16, s);
}

// ---------------------------------------------------------------------------
// fp32 -> bf16 convert (vectorized by 4)
// ---------------------------------------------------------------------------
__global__ void cvt_bf16(const float* __restrict__ in, __bf16* __restrict__ out, int n) {
    int i = (blockIdx.x * blockDim.x + threadIdx.x) * 4;
    if (i + 3 < n) {
        float4 f = *(const float4*)(in + i);
        out[i + 0] = f2bf(f.x);
        out[i + 1] = f2bf(f.y);
        out[i + 2] = f2bf(f.z);
        out[i + 3] = f2bf(f.w);
    }
}

// ---------------------------------------------------------------------------
// Transpose + convert the four 1024x1024 weights: Wt[n][k] = (bf16)W[k][n]
// ---------------------------------------------------------------------------
__global__ void transpose_cvt(const float* __restrict__ W0, const float* __restrict__ W1,
                              const float* __restrict__ W2, const float* __restrict__ W3,
                              __bf16* __restrict__ T0, __bf16* __restrict__ T1,
                              __bf16* __restrict__ T2, __bf16* __restrict__ T3) {
    const float* W; __bf16* T;
    switch (blockIdx.z) {
        case 0:  W = W0; T = T0; break;
        case 1:  W = W1; T = T1; break;
        case 2:  W = W2; T = T2; break;
        default: W = W3; T = T3; break;
    }
    __shared__ float tile[32][33];
    const int tx = threadIdx.x, ty = threadIdx.y;
    const int x0 = blockIdx.x * 32, y0 = blockIdx.y * 32;
    #pragma unroll
    for (int j = 0; j < 32; j += 8)
        tile[ty + j][tx] = W[(size_t)(y0 + ty + j) * 1024 + x0 + tx];
    __syncthreads();
    #pragma unroll
    for (int j = 0; j < 32; j += 8)
        T[(size_t)(x0 + ty + j) * 1024 + y0 + tx] = f2bf(tile[tx][ty + j]);
}

// ---------------------------------------------------------------------------
// bf16 WMMA GEMM: C[4096,1024] = A[4096,1024] x W (W given transposed Bt[n][k])
// One wave computes a 16(M) x 64(N) tile, K in steps of 32, register
// double-buffered so global loads overlap the WMMA pipe.
// mode 0: out bf16 row-major; mode 1: bf16 V-transposed [b][h][d][s]; mode 2: f32
// ---------------------------------------------------------------------------
__global__ __launch_bounds__(256) void gemm_bf16(
    const __bf16* __restrict__ A, const __bf16* __restrict__ Bt,
    const float* __restrict__ bias, void* __restrict__ outp, int mode) {
    const int lane = threadIdx.x & 31;
    const int wid  = blockIdx.x * 8 + (threadIdx.x >> 5);
    const int m0   = (wid >> 4) << 4;
    const int n0   = (wid & 15) << 6;
    const int al = lane & 15, ah = lane >> 4;

    const __bf16* arow = A  + (size_t)(m0 + al) * 1024 + ah * 16;
    const __bf16* brow = Bt + (size_t)(n0 + al) * 1024 + ah * 16;

    v8f acc[4] = {};
    V16 a0, a1;
    v16bf b0[4], b1[4];

    a0.h[0] = *(const v8bf*)(arow);
    a0.h[1] = *(const v8bf*)(arow + 8);
    #pragma unroll
    for (int t = 0; t < 4; t++) b0[t] = *(const v16bf*)(brow + (size_t)t * 16 * 1024);

    for (int k0 = 0; k0 < 1024; k0 += 64) {
        __builtin_prefetch((const void*)(arow + k0 + 128), 0, 3);
        // stage k0+32 while k0 computes
        a1.h[0] = *(const v8bf*)(arow + k0 + 32);
        a1.h[1] = *(const v8bf*)(arow + k0 + 40);
        #pragma unroll
        for (int t = 0; t < 4; t++)
            b1[t] = *(const v16bf*)(brow + (size_t)t * 16 * 1024 + k0 + 32);
        #pragma unroll
        for (int t = 0; t < 4; t++)
            acc[t] = __builtin_amdgcn_wmma_f32_16x16x32_bf16(
                false, a0.v, false, b0[t], (short)0, acc[t], false, false);
        // stage k0+64 while k0+32 computes
        if (k0 + 64 < 1024) {
            a0.h[0] = *(const v8bf*)(arow + k0 + 64);
            a0.h[1] = *(const v8bf*)(arow + k0 + 72);
            #pragma unroll
            for (int t = 0; t < 4; t++)
                b0[t] = *(const v16bf*)(brow + (size_t)t * 16 * 1024 + k0 + 64);
        }
        #pragma unroll
        for (int t = 0; t < 4; t++)
            acc[t] = __builtin_amdgcn_wmma_f32_16x16x32_bf16(
                false, a1.v, false, b1[t], (short)0, acc[t], false, false);
    }

    #pragma unroll
    for (int t = 0; t < 4; t++) {
        const int n = n0 + t * 16 + al;
        const float bv = bias[n];
        #pragma unroll
        for (int r = 0; r < 8; r++) {
            const int m = m0 + r + ah * 8;
            float c = acc[t][r] + bv;
            if (mode == 0) {
                ((__bf16*)outp)[(size_t)m * 1024 + n] = f2bf(c);
            } else if (mode == 1) {
                const int bb = m >> 11, s = m & 2047;
                const int hh = n >> 6,  d = n & 63;
                ((__bf16*)outp)[(((size_t)(bb * 16 + hh)) * 64 + d) * 2048 + s] = f2bf(c);
            } else {
                ((float*)outp)[(size_t)m * 1024 + n] = c;
            }
        }
    }
}

// ---------------------------------------------------------------------------
// Flash attention per (b,h): one wave = 16 query rows, streams 64-key blocks.
// All 4 waves of a block share (b,h): K-tile (64x64) and V-tile (64x64) are
// staged once per block into LDS with global_load_async_to_lds_b128
// (ASYNCcnt), then fed to WMMA from the DS pipe.
// Qb,Kb: bf16 [4096][1024];  Vt: bf16 [b][h][64][2048]
// ---------------------------------------------------------------------------
__global__ __launch_bounds__(128) void attn_kernel(
    const __bf16* __restrict__ Qb, const __bf16* __restrict__ Kb,
    const __bf16* __restrict__ Vt, const int* __restrict__ mask,
    float* __restrict__ scores_out, __bf16* __restrict__ Cb) {
    __shared__ __align__(32) __bf16 klds[64 * 64];     // [key][d]
    __shared__ __align__(32) __bf16 vlds[64 * 64];     // [d][key]
    __shared__ __align__(32) __bf16 pbuf[4][16 * 64];  // per-wave P bounce

    const int tid  = threadIdx.x;
    const int lane = tid & 31;
    const int wv   = tid >> 5;           // 0..3
    const int bh   = blockIdx.x >> 5;    // 0..31
    const int qgrp = blockIdx.x & 31;    // 0..31
    const int b = bh >> 4, h = bh & 15;
    const int q0 = (qgrp * 4 + wv) * 16;
    const int al = lane & 15, ah = lane >> 4;

    // Q A-tiles for the two 32-wide K-steps over D=64 (loaded once)
    V16 aq0, aq1;
    {
        const __bf16* qrow = Qb + (size_t)(b * 2048 + q0 + al) * 1024 + h * 64 + ah * 16;
        aq0.h[0] = *(const v8bf*)(qrow);
        aq0.h[1] = *(const v8bf*)(qrow + 8);
        aq1.h[0] = *(const v8bf*)(qrow + 32);
        aq1.h[1] = *(const v8bf*)(qrow + 40);
    }

    v8f ctx[4] = {};
    float mrow[8], lrow[8];
    #pragma unroll
    for (int r = 0; r < 8; r++) { mrow[r] = -3.0e38f; lrow[r] = 0.f; }

    const size_t vbase = ((size_t)(b * 16 + h)) * 64 * 2048;
    const size_t mbase = ((size_t)b * 2048) * 2048;
    const size_t sbase = (((size_t)(b * 16 + h)) * 2048) * 2048;
    __bf16* pl = pbuf[wv];

    // staging geometry: wave wv stages rows [wv*16, wv*16+16); each pass does
    // 4 rows (8 lanes x 16B per row)
    const int srow = wv * 16 + (lane >> 3);
    const int scol = (lane & 7) * 8;   // element column (8 bf16 = 16B)
    const unsigned lk = (unsigned)(size_t)(&klds[srow * 64 + scol]);
    const unsigned lv = (unsigned)(size_t)(&vlds[srow * 64 + scol]);

    for (int key0 = 0; key0 < 2048; key0 += 64) {
        __syncthreads();   // previous iteration's LDS consumers are done
        {
            const __bf16* gk = Kb + (size_t)(b * 2048 + key0 + srow) * 1024 + h * 64 + scol;
            const __bf16* gv = Vt + vbase + (size_t)srow * 2048 + key0 + scol;
            #pragma unroll
            for (int i = 0; i < 4; i++) {
                asm volatile("global_load_async_to_lds_b128 %0, %1, off"
                             :: "v"(lk + i * 4 * 128), "v"((const void*)(gk + i * 4 * 1024))
                             : "memory");
                asm volatile("global_load_async_to_lds_b128 %0, %1, off"
                             :: "v"(lv + i * 4 * 128), "v"((const void*)(gv + i * 4 * 2048))
                             : "memory");
            }
            asm volatile("s_wait_asynccnt 0" ::: "memory");
        }
        __syncthreads();   // tiles visible to all 4 waves

        // ---- S = Q K^T for 16q x 64k block (K-tiles from LDS) ----
        v8f sacc[4] = {};
        #pragma unroll
        for (int ks = 0; ks < 2; ks++) {
            v16bf a = ks ? aq1.v : aq0.v;
            #pragma unroll
            for (int t = 0; t < 4; t++) {
                v16bf bt = *(const v16bf*)(&klds[(t * 16 + al) * 64 + ks * 32 + ah * 16]);
                sacc[t] = __builtin_amdgcn_wmma_f32_16x16x32_bf16(
                    false, a, false, bt, (short)0, sacc[t], false, false);
            }
        }
        // ---- scale, mask (NT load), emit scores (NT store) ----
        #pragma unroll
        for (int t = 0; t < 4; t++) {
            const int key = key0 + t * 16 + al;
            #pragma unroll
            for (int r = 0; r < 8; r++) {
                const int q = q0 + r + ah * 8;
                float s = sacc[t][r] * 0.125f;
                int mv = __builtin_nontemporal_load(&mask[mbase + (size_t)q * 2048 + key]);
                if (mv == 0) s = -1.0e9f;
                sacc[t][r] = s;
                __builtin_nontemporal_store(s, &scores_out[sbase + (size_t)q * 2048 + key]);
            }
        }
        // ---- online softmax (row stats over 16 lanes per half-wave) ----
        #pragma unroll
        for (int r = 0; r < 8; r++) {
            float mn = fmaxf(fmaxf(mrow[r], sacc[0][r]),
                             fmaxf(fmaxf(sacc[1][r], sacc[2][r]), sacc[3][r]));
            mn = fmaxf(mn, __shfl_xor(mn, 1, 32));
            mn = fmaxf(mn, __shfl_xor(mn, 2, 32));
            mn = fmaxf(mn, __shfl_xor(mn, 4, 32));
            mn = fmaxf(mn, __shfl_xor(mn, 8, 32));
            const float sc = __expf(mrow[r] - mn);
            mrow[r] = mn;
            float ls = 0.f;
            #pragma unroll
            for (int t = 0; t < 4; t++) {
                float p = __expf(sacc[t][r] - mn);
                sacc[t][r] = p;
                ls += p;
            }
            ls += __shfl_xor(ls, 1, 32);
            ls += __shfl_xor(ls, 2, 32);
            ls += __shfl_xor(ls, 4, 32);
            ls += __shfl_xor(ls, 8, 32);
            lrow[r] = lrow[r] * sc + ls;
            ctx[0][r] *= sc; ctx[1][r] *= sc; ctx[2][r] *= sc; ctx[3][r] *= sc;
        }
        // ---- bounce P through LDS: C/D layout -> A layout (bf16) ----
        asm volatile("s_wait_dscnt 0" ::: "memory");
        #pragma unroll
        for (int t = 0; t < 4; t++)
            #pragma unroll
            for (int r = 0; r < 8; r++)
                pl[(r + ah * 8) * 64 + t * 16 + al] = f2bf(sacc[t][r]);
        asm volatile("s_wait_dscnt 0" ::: "memory");
        // ---- ctx += P V (V-tiles from LDS) ----
        #pragma unroll
        for (int kb = 0; kb < 2; kb++) {
            V16 ap;
            const __bf16* prow = pl + al * 64 + kb * 32 + ah * 16;
            ap.h[0] = *(const v8bf*)(prow);
            ap.h[1] = *(const v8bf*)(prow + 8);
            #pragma unroll
            for (int t = 0; t < 4; t++) {
                v16bf bv = *(const v16bf*)(&vlds[(t * 16 + al) * 64 + kb * 32 + ah * 16]);
                ctx[t] = __builtin_amdgcn_wmma_f32_16x16x32_bf16(
                    false, ap.v, false, bv, (short)0, ctx[t], false, false);
            }
        }
    }

    // ---- normalize and store ctx (bf16, [b*2048+q][h*64+d]) ----
    #pragma unroll
    for (int t = 0; t < 4; t++)
        #pragma unroll
        for (int r = 0; r < 8; r++) {
            const int q = q0 + r + ah * 8;
            float v = ctx[t][r] / lrow[r];
            Cb[(size_t)(b * 2048 + q) * 1024 + h * 64 + t * 16 + al] = f2bf(v);
        }
}

// ---------------------------------------------------------------------------
// Residual + LayerNorm: out = LN(Hf + x) * g + b, one block per row
// ---------------------------------------------------------------------------
__global__ __launch_bounds__(256) void ln_kernel(
    const float* __restrict__ Hf, const float* __restrict__ x,
    const float* __restrict__ g, const float* __restrict__ bta,
    float* __restrict__ out) {
    __shared__ float red[256];
    const int row = blockIdx.x;
    const int tid = threadIdx.x;
    float v[4];
    float s = 0.f;
    #pragma unroll
    for (int j = 0; j < 4; j++) {
        const int idx = j * 256 + tid;
        float y = Hf[(size_t)row * 1024 + idx] + x[(size_t)row * 1024 + idx];
        v[j] = y; s += y;
    }
    red[tid] = s; __syncthreads();
    for (int o = 128; o > 0; o >>= 1) {
        if (tid < o) red[tid] += red[tid + o];
        __syncthreads();
    }
    const float mu = red[0] * (1.f / 1024.f);
    __syncthreads();
    float s2 = 0.f;
    #pragma unroll
    for (int j = 0; j < 4; j++) { float d = v[j] - mu; s2 += d * d; }
    red[tid] = s2; __syncthreads();
    for (int o = 128; o > 0; o >>= 1) {
        if (tid < o) red[tid] += red[tid + o];
        __syncthreads();
    }
    const float rstd = rsqrtf(red[0] * (1.f / 1024.f) + 1e-12f);
    #pragma unroll
    for (int j = 0; j < 4; j++) {
        const int idx = j * 256 + tid;
        out[(size_t)row * 1024 + idx] = (v[j] - mu) * rstd * g[idx] + bta[idx];
    }
}

// ---------------------------------------------------------------------------
extern "C" void kernel_launch(void* const* d_in, const int* in_sizes, int n_in,
                              void* d_out, int out_size, void* d_ws, size_t ws_size,
                              hipStream_t stream) {
    const float* x    = (const float*)d_in[0];
    const int*   mask = (const int*)  d_in[1];
    const float* Wq   = (const float*)d_in[2];
    const float* bq   = (const float*)d_in[3];
    const float* Wk   = (const float*)d_in[4];
    const float* bk   = (const float*)d_in[5];
    const float* Wv   = (const float*)d_in[6];
    const float* bv   = (const float*)d_in[7];
    const float* Wo   = (const float*)d_in[8];
    const float* bo   = (const float*)d_in[9];
    const float* lng  = (const float*)d_in[10];
    const float* lnb  = (const float*)d_in[11];

    float* out    = (float*)d_out;                    // [2,2048,1024]
    float* scores = out + (size_t)2 * 2048 * 1024;    // [2,16,2048,2048]

    char* ws = (char*)d_ws;
    const size_t MB = 1024 * 1024;
    __bf16* Xb  = (__bf16*)(ws + 0);        // 8 MB
    __bf16* Wqt = (__bf16*)(ws + 8  * MB);  // 2 MB each
    __bf16* Wkt = (__bf16*)(ws + 10 * MB);
    __bf16* Wvt = (__bf16*)(ws + 12 * MB);
    __bf16* Wot = (__bf16*)(ws + 14 * MB);
    __bf16* Qb  = (__bf16*)(ws + 16 * MB);  // 8 MB
    __bf16* Kb  = (__bf16*)(ws + 24 * MB);  // 8 MB
    __bf16* Vt  = (__bf16*)(ws + 32 * MB);  // 8 MB
    __bf16* Cb  = (__bf16*)(ws + 40 * MB);  // 8 MB
    float*  Hf  = (float*) (ws + 16 * MB);  // 16 MB, aliases dead Qb/Kb

    cvt_bf16<<<4096, 256, 0, stream>>>(x, Xb, 4096 * 1024);
    dim3 tb(32, 8), tg(32, 32, 4);
    transpose_cvt<<<tg, tb, 0, stream>>>(Wq, Wk, Wv, Wo, Wqt, Wkt, Wvt, Wot);

    gemm_bf16<<<512, 256, 0, stream>>>(Xb, Wqt, bq, Qb, 0);
    gemm_bf16<<<512, 256, 0, stream>>>(Xb, Wkt, bk, Kb, 0);
    gemm_bf16<<<512, 256, 0, stream>>>(Xb, Wvt, bv, Vt, 1);

    attn_kernel<<<1024, 128, 0, stream>>>(Qb, Kb, Vt, mask, scores, Cb);

    gemm_bf16<<<512, 256, 0, stream>>>(Cb, Wot, bo, Hf, 2);
    ln_kernel<<<4096, 256, 0, stream>>>(Hf, x, lng, lnb, out);
}
